// LSTMRegressor_16947940950333
// MI455X (gfx1250) — matrix operand order repaired and checked by reading, not compiled, not run on hardware
//
#include <hip/hip_runtime.h>
#include <hip/hip_bf16.h>
#include <math.h>

typedef __attribute__((ext_vector_type(16))) _Float16 v16h;
typedef __attribute__((ext_vector_type(8)))  _Float16 v8h;
typedef __attribute__((ext_vector_type(8)))  float    v8f;

#define B_TOT   64
#define T_LEN   512
#define IN_DIM  64
#define H_DIM   512
#define G_DIM   2048          // 4*H
#define NT_TOT  128           // G_DIM / 16
#define KT0     18            // (64 + 512) / 32
#define KT1     32            // (512 + 512) / 32
#define AP      1032          // LDS A-buffer pitch in halves (1024 + 8 pad)

#define WMMA(A, Bm, C) __builtin_amdgcn_wmma_f32_16x16x32_f16( \
    false, (A), false, (Bm), (short)0, (C), false, false)

__device__ __forceinline__ float fast_tanhf(float x) {
#if __has_builtin(__builtin_amdgcn_tanhf)
    return __builtin_amdgcn_tanhf(x);          // v_tanh_f32 (gfx1250 TRANS op)
#else
    return tanhf(x);
#endif
}
__device__ __forceinline__ float fast_sigmoid(float x) {
    return __builtin_fmaf(0.5f, fast_tanhf(0.5f * x), 0.5f);
}

// ---------------------------------------------------------------------------
// Pre-pack combined [W_ih ; W_hh] into the exact WMMA B-fragment layout:
// dst[((nt*KT + kt)*32 + lane)*16 + h] = W[n = nt*16 + lane%16,
//                                         k = kt*32 + (h/8)*16 + (lane/16)*8 + h%8] (f16)
// ---------------------------------------------------------------------------
__global__ void pack_w_kernel(const float* __restrict__ w_ih,
                              const float* __restrict__ w_hh,
                              _Float16* __restrict__ dst, int KT, int inK)
{
    int idx   = blockIdx.x * 256 + threadIdx.x;
    int total = NT_TOT * KT * 512;
    if (idx >= total) return;
    int h    = idx & 15;
    int lane = (idx >> 4) & 31;
    int rest = idx >> 9;
    int kt   = rest % KT;
    int nt   = rest / KT;
    int n = nt * 16 + (lane & 15);
    int k = kt * 32 + ((h >> 3) << 4) + ((lane >> 4) << 3) + (h & 7);
    float v = (k < inK) ? w_ih[n * inK + k] : w_hh[n * H_DIM + (k - inK)];
    dst[idx] = (_Float16)v;
}

__global__ void cvt_f16_kernel(const float* __restrict__ src,
                               _Float16* __restrict__ dst, int n)
{
    int i = blockIdx.x * 256 + threadIdx.x;
    if (i < n) dst[i] = (_Float16)src[i];
}

// ---------------------------------------------------------------------------
// One LSTM layer, fused input+recurrent GEMM, software-pipelined WMMA.
// Workgroup = 16 batch rows, 512 threads = 16 waves; wave owns 8 N-tiles.
// ---------------------------------------------------------------------------
template<int KT, int INK, bool FIRST>
__device__ __forceinline__ void run_layer(
    _Float16* sa, float* gates,
    const _Float16* __restrict__ wp,
    const float* __restrict__ b_ih, const float* __restrict__ b_hh,
    const _Float16* __restrict__ xin,   // FIRST: x (f16, [B][T][64]); else h1 ([T][B][512])
    _Float16* __restrict__ h1g,         // FIRST: layer-0 output
    float* pool)                        // !FIRST: pooling accumulator
{
    const int tid  = threadIdx.x;
    const int lane = tid & 31;
    const int wave = tid >> 5;
    const int m    = lane & 15;         // A row / B,C column within tile
    const int hi   = lane >> 4;
    const int b0   = blockIdx.x * 16;

    // per-thread gate biases (constant over t) — registers, no LDS
    const float bi = b_ih[tid]        + b_hh[tid];
    const float bf = b_ih[512 + tid]  + b_hh[512 + tid];
    const float bg = b_ih[1024 + tid] + b_hh[1024 + tid];
    const float bo = b_ih[1536 + tid] + b_hh[1536 + tid];

    // zero h region of A-buffer (cols [INK, INK+512))
    {
        int r = tid >> 5, c0 = lane * 16;
        uint4 z = {0u, 0u, 0u, 0u};
        uint4* zd = (uint4*)(sa + r * AP + INK + c0);
        zd[0] = z; zd[1] = z;
    }
    float c_reg[16];
#pragma unroll
    for (int i = 0; i < 16; ++i) c_reg[i] = 0.0f;
    __syncthreads();

    // per-wave B base: immediate offsets cover j; bump 512 halves per kt
    const _Float16* wb0   = wp + ((size_t)wave * 8 * KT) * 512 + lane * 16;
    const _Float16* arow0 = sa + m * AP + hi * 8;
    // single-base gates scatter: byte offset (v*8192 + j*64) fits 16-bit DS imm
    float* gbase = gates + (hi * 8) * G_DIM + wave * 128 + m;

    for (int t = 0; t < T_LEN; ++t) {
        // ---- stage this step's input into A-buffer cols [0, INK)
        if (FIRST) {
            int r = tid >> 5;
            const unsigned* src =
                (const unsigned*)(xin + ((size_t)(b0 + r) * T_LEN + t) * IN_DIM);
            *(unsigned*)(sa + r * AP + 2 * lane) = src[lane];
        } else {
            int r = tid >> 5, c0 = lane * 16;
            const uint4* src =
                (const uint4*)(xin + ((size_t)t * B_TOT + b0 + r) * H_DIM + c0);
            uint4* d = (uint4*)(sa + r * AP + c0);
            d[0] = src[0]; d[1] = src[1];
        }
        __syncthreads();

        // ---- WMMA GEMM: gates(16x2048) = A(16xK) * Wpack(Kx2048)
        v8f acc[8] = {};
        const _Float16* wb = wb0;

        auto LB = [&](const _Float16* p, int j) -> v16h {
            return *(const v16h*)(p + (size_t)j * (KT * 512));   // imm offset per j
        };
        auto LA = [&](int kt) -> v16h {
            union { v16h v; v8h p[2]; } u;
            u.p[0] = *(const v8h*)(arow0 + kt * 32);        // K = hi*8 + 0..7
            u.p[1] = *(const v8h*)(arow0 + kt * 32 + 16);   // K = 16 + hi*8 + 0..7
            return u.v;
        };

        // prologue: first half-group of kt=0 in flight
        v16h bq0 = LB(wb, 0), bq1 = LB(wb, 1), bq2 = LB(wb, 2), bq3 = LB(wb, 3);

        for (int kt = 0; kt < KT - 1; ++kt, wb += 512) {
            v16h a = LA(kt);
            // issue second half-group of this kt before any WMMA
            v16h b4 = LB(wb, 4), b5 = LB(wb, 5), b6 = LB(wb, 6), b7 = LB(wb, 7);
            acc[0] = WMMA(a, bq0, acc[0]);
            acc[1] = WMMA(a, bq1, acc[1]);
            acc[2] = WMMA(a, bq2, acc[2]);
            acc[3] = WMMA(a, bq3, acc[3]);
            // issue first half-group of kt+1 behind these WMMAs
            bq0 = LB(wb + 512, 0); bq1 = LB(wb + 512, 1);
            bq2 = LB(wb + 512, 2); bq3 = LB(wb + 512, 3);
            acc[4] = WMMA(a, b4, acc[4]);
            acc[5] = WMMA(a, b5, acc[5]);
            acc[6] = WMMA(a, b6, acc[6]);
            acc[7] = WMMA(a, b7, acc[7]);
        }
        {   // epilogue kt = KT-1
            v16h a  = LA(KT - 1);
            v16h b4 = LB(wb, 4), b5 = LB(wb, 5), b6 = LB(wb, 6), b7 = LB(wb, 7);
            acc[0] = WMMA(a, bq0, acc[0]);
            acc[1] = WMMA(a, bq1, acc[1]);
            acc[2] = WMMA(a, bq2, acc[2]);
            acc[3] = WMMA(a, bq3, acc[3]);
            acc[4] = WMMA(a, b4, acc[4]);
            acc[5] = WMMA(a, b5, acc[5]);
            acc[6] = WMMA(a, b6, acc[6]);
            acc[7] = WMMA(a, b7, acc[7]);
        }

        // scatter C frags off ONE base VGPR with compile-time DS offsets:
        // (v + 8*hi) row stride 8192 B, j column-group stride 64 B (max 57792 B)
#pragma unroll
        for (int j = 0; j < 8; ++j) {
#pragma unroll
            for (int v = 0; v < 8; ++v)
                gbase[v * G_DIM + j * 16] = acc[j][v];
        }
        __syncthreads();

        // ---- activations: thread owns hidden column tid, rows 0..15
#pragma unroll
        for (int r = 0; r < 16; ++r) {
            float pi = gates[r * G_DIM + tid]        + bi;
            float pf = gates[r * G_DIM + 512 + tid]  + bf;
            float pg = gates[r * G_DIM + 1024 + tid] + bg;
            float po = gates[r * G_DIM + 1536 + tid] + bo;
            float ig = fast_sigmoid(pi);
            float fg = fast_sigmoid(pf);
            float gg = fast_tanhf(pg);
            float og = fast_sigmoid(po);
            float c  = fg * c_reg[r] + ig * gg;
            c_reg[r] = c;
            float h  = og * fast_tanhf(c);
            sa[r * AP + INK + tid] = (_Float16)h;        // feed next step
            if (FIRST)
                h1g[((size_t)t * B_TOT + b0 + r) * H_DIM + tid] = (_Float16)h;
            else
                pool[r] += h;
        }
        __syncthreads();
    }
}

__global__ void __launch_bounds__(512, 1)
lstm_kernel(const _Float16* __restrict__ xf16,
            const _Float16* __restrict__ w0,
            const _Float16* __restrict__ w1,
            const float* __restrict__ bi0, const float* __restrict__ bh0,
            const float* __restrict__ bi1, const float* __restrict__ bh1,
            _Float16* __restrict__ h1g,
            float* __restrict__ pooled)
{
    extern __shared__ char smem_raw[];
    _Float16* sa    = (_Float16*)smem_raw;              // 16 x AP halves
    float*    gates = (float*)(smem_raw + 16 * AP * 2); // 16 x 2048 f32

    float pool[16];
#pragma unroll
    for (int i = 0; i < 16; ++i) pool[i] = 0.0f;

    run_layer<KT0, IN_DIM, true >(sa, gates, w0, bi0, bh0, xf16, h1g, pool);
    run_layer<KT1, H_DIM, false>(sa, gates, w1, bi1, bh1, h1g, (_Float16*)nullptr, pool);

    const int tid = threadIdx.x;
    const int b0  = blockIdx.x * 16;
#pragma unroll
    for (int r = 0; r < 16; ++r)
        pooled[(size_t)(b0 + r) * H_DIM + tid] = pool[r] * (1.0f / (float)T_LEN);
}

// ---------------------------------------------------------------------------
// Head: z = relu(pooled @ fc1_w^T + fc1_b); out = z @ fc_w^T + fc_b  (tiny)
// ---------------------------------------------------------------------------
__global__ void fc1_kernel(const float* __restrict__ pooled,
                           const float* __restrict__ w,
                           const float* __restrict__ b,
                           float* __restrict__ z)
{
    int mcol = threadIdx.x;      // 0..255
    int bb   = blockIdx.x;       // 0..63
    const float* pr = pooled + (size_t)bb * H_DIM;
    const float* wr = w + (size_t)mcol * H_DIM;
    float s = 0.0f;
    for (int k = 0; k < H_DIM; ++k) s += pr[k] * wr[k];
    s += b[mcol];
    z[bb * 256 + mcol] = fmaxf(s, 0.0f);
}

__global__ void fc2_kernel(const float* __restrict__ z,
                           const float* __restrict__ w,
                           const float* __restrict__ b,
                           float* __restrict__ out)
{
    int idx = threadIdx.x;       // 0..511
    int bb  = idx >> 3;
    int o   = idx & 7;
    const float* zr = z + bb * 256;
    const float* wr = w + o * 256;
    float s = 0.0f;
    for (int k = 0; k < 256; ++k) s += zr[k] * wr[k];
    out[idx] = s + b[o];
}

// ---------------------------------------------------------------------------
extern "C" void kernel_launch(void* const* d_in, const int* in_sizes, int n_in,
                              void* d_out, int out_size, void* d_ws, size_t ws_size,
                              hipStream_t stream)
{
    const float* x     = (const float*)d_in[0];
    const float* W_ih0 = (const float*)d_in[1];
    const float* W_hh0 = (const float*)d_in[2];
    const float* b_ih0 = (const float*)d_in[3];
    const float* b_hh0 = (const float*)d_in[4];
    const float* W_ih1 = (const float*)d_in[5];
    const float* W_hh1 = (const float*)d_in[6];
    const float* b_ih1 = (const float*)d_in[7];
    const float* b_hh1 = (const float*)d_in[8];
    const float* fc1_w = (const float*)d_in[9];
    const float* fc1_b = (const float*)d_in[10];
    const float* fc_w  = (const float*)d_in[11];
    const float* fc_b  = (const float*)d_in[12];

    char* ws = (char*)d_ws;
    size_t off = 0;
    auto carve = [&](size_t bytes) { void* p = ws + off; off += (bytes + 255) & ~size_t(255); return p; };
    _Float16* w0p    = (_Float16*)carve((size_t)NT_TOT * KT0 * 512 * 2);     // 2.36 MB
    _Float16* w1p    = (_Float16*)carve((size_t)NT_TOT * KT1 * 512 * 2);     // 4.19 MB
    _Float16* xf     = (_Float16*)carve((size_t)B_TOT * T_LEN * IN_DIM * 2); // 4.19 MB
    _Float16* h1g    = (_Float16*)carve((size_t)T_LEN * B_TOT * H_DIM * 2);  // 33.6 MB
    float*    pooled = (float*)   carve((size_t)B_TOT * H_DIM * 4);
    float*    zbuf   = (float*)   carve((size_t)B_TOT * 256 * 4);
    (void)ws_size; (void)in_sizes; (void)n_in; (void)out_size;

    int t0 = NT_TOT * KT0 * 512;
    pack_w_kernel<<<(t0 + 255) / 256, 256, 0, stream>>>(W_ih0, W_hh0, w0p, KT0, IN_DIM);
    int t1 = NT_TOT * KT1 * 512;
    pack_w_kernel<<<(t1 + 255) / 256, 256, 0, stream>>>(W_ih1, W_hh1, w1p, KT1, H_DIM);
    int nx = B_TOT * T_LEN * IN_DIM;
    cvt_f16_kernel<<<(nx + 255) / 256, 256, 0, stream>>>(x, xf, nx);

    size_t smem = (size_t)16 * AP * 2 + (size_t)16 * G_DIM * 4;   // 164096 B (<320KB/WGP)
    hipFuncSetAttribute((const void*)lstm_kernel,
                        hipFuncAttributeMaxDynamicSharedMemorySize, (int)smem);
    lstm_kernel<<<4, 512, smem, stream>>>(xf, w0p, w1p, b_ih0, b_hh0, b_ih1, b_hh1,
                                          h1g, pooled);

    fc1_kernel<<<B_TOT, 256, 0, stream>>>(pooled, fc1_w, fc1_b, zbuf);
    fc2_kernel<<<1, 512, 0, stream>>>(zbuf, fc_w, fc_b, (float*)d_out);
}
